// ValueHead_24455543783959
// MI455X (gfx1250) — matrix operand order repaired and checked by reading, not compiled, not run on hardware
//
#include <hip/hip_runtime.h>
#include <hip/hip_bf16.h>

typedef __attribute__((ext_vector_type(16))) _Float16 v16h;
typedef __attribute__((ext_vector_type(8)))  float    v8f;

#define C_IN 256
#define CH_O 32
#define HW   361
#define NB   1024
#define NTILES 23   // ceil(361/16)

// ---------------------------------------------------------------------------
// Kernel 0: repack conv_w [32,256] f32 -> f16 A-fragments.
// Layout: aF[((mtile*8 + kstep)*32 + lane)*16 + (2*v + h)]
// 16-bit A 16x32 table: lane%16 = M; v in 0..3 -> K = 32*kstep + (lane<16?0:8) + 2v + h
//                        v in 4..7 -> K = 32*kstep + 16 + (lane<16?0:8) + 2(v-4) + h
// ---------------------------------------------------------------------------
__global__ void prep_conv_w(const float* __restrict__ conv_w, _Float16* __restrict__ aF) {
    for (int idx = threadIdx.x; idx < 2 * 8 * 32 * 16; idx += blockDim.x) {
        int h     = idx & 1;
        int v     = (idx >> 1) & 7;
        int lane  = (idx >> 4) & 31;
        int kstep = (idx >> 9) & 7;
        int mtile = (idx >> 12) & 1;
        int M  = 16 * mtile + (lane & 15);
        int kb = (lane < 16) ? 0 : 8;
        int K  = (v < 4) ? (32 * kstep + kb + 2 * v + h)
                         : (32 * kstep + 16 + kb + 2 * (v - 4) + h);
        aF[idx] = (_Float16)conv_w[M * C_IN + K];
    }
}

// ---------------------------------------------------------------------------
// Kernel 1: per-batch conv (WMMA f16 -> f32 acc) fused with global mean/max
// pooling. One block = one batch, 8 waves; wave w handles N-tiles w, w+8, w+16.
// B fragment layout (16-bit KxN=32x16): lane%16 = n, k in the same table order as A:
//   k = 32*kstep + (lane<16 ? 0 : 8) + elem + (elem<8 ? 0 : 8)   (elem = 0..15)
// ---------------------------------------------------------------------------
__global__ void __launch_bounds__(256) conv_pool(const float* __restrict__ x,
                                                 const _Float16* __restrict__ aF,
                                                 float* __restrict__ Vp) {
    __shared__ _Float16 aLds[2 * 8 * 32 * 16];  // 16 KB
    __shared__ float lsum[8][32];
    __shared__ float lmax[8][32];

    const int b    = blockIdx.x;
    const int lane = threadIdx.x & 31;
    const int wv   = threadIdx.x >> 5;

    // stage A fragments into LDS (shared by all waves, resident in L2 anyway)
    for (int i = threadIdx.x; i < (2 * 8 * 32 * 16) / 2; i += blockDim.x)
        ((unsigned int*)aLds)[i] = ((const unsigned int*)aF)[i];
    __syncthreads();

    const float NEG_INF = -__builtin_inff();
    float sAcc0[8], sAcc1[8], mAcc0[8], mAcc1[8];
#pragma unroll
    for (int r = 0; r < 8; ++r) {
        sAcc0[r] = 0.f; sAcc1[r] = 0.f;
        mAcc0[r] = NEG_INF; mAcc1[r] = NEG_INF;
    }

    const int   koff = (lane < 16) ? 0 : 8;
    const float* xb0 = x + (size_t)b * (C_IN * HW);

    for (int nt = wv; nt < NTILES; nt += 8) {
        const int  s     = nt * 16 + (lane & 15);
        const bool valid = (s < HW);
        const float* xb  = xb0 + s;

        v8f c0 = {0.f,0.f,0.f,0.f,0.f,0.f,0.f,0.f};
        v8f c1 = {0.f,0.f,0.f,0.f,0.f,0.f,0.f,0.f};

#pragma unroll
        for (int ks = 0; ks < 8; ++ks) {
            const int kBase = ks * 32 + koff;
            v16h bf;
#pragma unroll
            for (int e = 0; e < 16; ++e) {
                float f = valid ? __builtin_nontemporal_load(&xb[(size_t)(kBase + e + (e >= 8 ? 8 : 0)) * HW])
                                : 0.f;
                bf[e] = (_Float16)f;
            }
            v16h a0 = *(const v16h*)&aLds[(ks * 32 + lane) * 16];
            v16h a1 = *(const v16h*)&aLds[((8 + ks) * 32 + lane) * 16];
            c0 = __builtin_amdgcn_wmma_f32_16x16x32_f16(false, a0, false, bf,
                                                        (short)0, c0, false, false);
            c1 = __builtin_amdgcn_wmma_f32_16x16x32_f16(false, a1, false, bf,
                                                        (short)0, c1, false, false);
        }
#pragma unroll
        for (int r = 0; r < 8; ++r) {
            float v0 = c0[r], v1 = c1[r];
            sAcc0[r] += v0;  // invalid columns produced exactly 0
            sAcc1[r] += v1;
            if (valid) {
                mAcc0[r] = fmaxf(mAcc0[r], v0);
                mAcc1[r] = fmaxf(mAcc1[r], v1);
            }
        }
    }

    // reduce across the 16 lanes of each half-group (xor masks stay in-group)
#pragma unroll
    for (int r = 0; r < 8; ++r) {
        float s0 = sAcc0[r], s1 = sAcc1[r], m0 = mAcc0[r], m1 = mAcc1[r];
#pragma unroll
        for (int msk = 8; msk >= 1; msk >>= 1) {
            s0 += __shfl_xor(s0, msk, 32);
            s1 += __shfl_xor(s1, msk, 32);
            m0 = fmaxf(m0, __shfl_xor(m0, msk, 32));
            m1 = fmaxf(m1, __shfl_xor(m1, msk, 32));
        }
        if ((lane & 15) == 0) {
            int off = (lane >> 4) * 8;  // lanes 16..31 carry M = r+8
            lsum[wv][off + r]      = s0;  lmax[wv][off + r]      = m0;
            lsum[wv][16 + off + r] = s1;  lmax[wv][16 + off + r] = m1;
        }
    }
    __syncthreads();

    if (threadIdx.x < 32) {
        int ch = threadIdx.x;
        float s = 0.f, m = NEG_INF;
#pragma unroll
        for (int ww = 0; ww < 8; ++ww) {
            s += lsum[ww][ch];
            m = fmaxf(m, lmax[ww][ch]);
        }
        float mean = s * (1.0f / 361.0f);
        float* vpb = Vp + (size_t)b * 96;
        vpb[ch]      = mean;
        vpb[32 + ch] = mean;   // (H - B_AVG)/10 = (19-9)/10 = 1.0
        vpb[64 + ch] = m;
    }
}

// ---------------------------------------------------------------------------
// Kernel 2: FC heads + softmax. One wave per batch.
// ---------------------------------------------------------------------------
__device__ __forceinline__ float wredSum(float v) {
#pragma unroll
    for (int m = 16; m >= 1; m >>= 1) v += __shfl_xor(v, m, 32);
    return v;
}
__device__ __forceinline__ float wredMax(float v) {
#pragma unroll
    for (int m = 16; m >= 1; m >>= 1) v = fmaxf(v, __shfl_xor(v, m, 32));
    return v;
}

__global__ void __launch_bounds__(256) heads_kernel(
        const float* __restrict__ Vp,
        const float* __restrict__ fc1_w, const float* __restrict__ fc1_b,
        const float* __restrict__ fc2_w, const float* __restrict__ fc2_b,
        const float* __restrict__ fc3_w, const float* __restrict__ fc3_b,
        const float* __restrict__ sc_w,  const float* __restrict__ sc_b,
        const float* __restrict__ fc4_w, const float* __restrict__ fc4_b,
        const float* __restrict__ fc5_w, const float* __restrict__ fc5_b,
        float* __restrict__ out) {
    const int lane = threadIdx.x & 31;
    const int wv   = threadIdx.x >> 5;
    const int b    = blockIdx.x * (blockDim.x >> 5) + wv;
    if (b >= NB) return;

    const float* vpb = Vp + (size_t)b * 96;
    float vp0 = vpb[lane], vp1 = vpb[32 + lane], vp2 = vpb[64 + lane];

    const bool hasB = (lane < 16);
    const int  jb   = hasB ? (lane + 32) : lane;  // second output row (masked later)

    float h1a = fc1_b[lane], h1b = fc1_b[jb];
    float h3a = fc3_b[lane], h3b = fc3_b[jb];
    float h4a = fc4_b[lane], h4b = fc4_b[jb];

    for (int k = 0; k < 96; ++k) {
        float v = (k < 32) ? __shfl(vp0, k, 32)
                : (k < 64) ? __shfl(vp1, k - 32, 32)
                           : __shfl(vp2, k - 64, 32);
        h1a += fc1_w[lane * 96 + k] * v;  h1b += fc1_w[jb * 96 + k] * v;
        h3a += fc3_w[lane * 96 + k] * v;  h3b += fc3_w[jb * 96 + k] * v;
        h4a += fc4_w[lane * 97 + k] * v;  h4b += fc4_w[jb * 97 + k] * v;
    }

    // game outcome head: tanh(relu(h1) @ fc2_w + fc2_b)
    float t = fmaxf(h1a, 0.f) * fc2_w[lane] + (hasB ? fmaxf(h1b, 0.f) * fc2_w[jb] : 0.f);
    float game = tanhf(wredSum(t) + fc2_b[0]);
    if (lane == 0) out[b] = game;

    // scaling head
    t = fmaxf(h3a, 0.f) * sc_w[lane] + (hasB ? fmaxf(h3b, 0.f) * sc_w[jb] : 0.f);
    float scal = wredSum(t) + sc_b[0];

    // distance logits: d index 0..40 <-> dval = d-20
    float dca = fc4_w[lane * 97 + 96], dcb = fc4_w[jb * 97 + 96];
    float w5a = fc5_w[lane],           w5b = fc5_w[jb];
    float b5  = fc5_b[0];
    float l1 = 0.f, l2 = 0.f;
    for (int d = 0; d < 41; ++d) {
        float dv = (float)(d - 20);
        float t2 = fmaxf(h4a + dv * dca, 0.f) * w5a
                 + (hasB ? fmaxf(h4b + dv * dcb, 0.f) * w5b : 0.f);
        float lg = (wredSum(t2) + b5) * scal;
        if (lane == d) l1 = lg;
        if (d >= 32 && lane == d - 32) l2 = lg;
    }

    // softmax over 41 (lane holds d=lane; lanes 0..8 also hold d=lane+32)
    const float NEG_INF = -__builtin_inff();
    float m = wredMax(fmaxf(l1, (lane < 9) ? l2 : NEG_INF));
    float e1 = expf(l1 - m);
    float e2 = (lane < 9) ? expf(l2 - m) : 0.f;
    float Z  = wredSum(e1 + e2);
    float inv = 1.f / Z;
    float* pr = out + NB + (size_t)b * 41;
    pr[lane] = e1 * inv;
    if (lane < 9) pr[32 + lane] = e2 * inv;
}

// ---------------------------------------------------------------------------
extern "C" void kernel_launch(void* const* d_in, const int* in_sizes, int n_in,
                              void* d_out, int out_size, void* d_ws, size_t ws_size,
                              hipStream_t stream) {
    const float* x      = (const float*)d_in[0];
    const float* conv_w = (const float*)d_in[1];
    const float* fc1_w  = (const float*)d_in[2];
    const float* fc1_b  = (const float*)d_in[3];
    const float* fc2_w  = (const float*)d_in[4];
    const float* fc2_b  = (const float*)d_in[5];
    const float* fc3_w  = (const float*)d_in[6];
    const float* fc3_b  = (const float*)d_in[7];
    const float* sc_w   = (const float*)d_in[8];
    const float* sc_b   = (const float*)d_in[9];
    const float* fc4_w  = (const float*)d_in[10];
    const float* fc4_b  = (const float*)d_in[11];
    const float* fc5_w  = (const float*)d_in[12];
    const float* fc5_b  = (const float*)d_in[13];

    _Float16* aF = (_Float16*)d_ws;                       // 16 KB
    float*    Vp = (float*)((char*)d_ws + 16384);         // 1024*96*4 = 384 KB

    prep_conv_w<<<1, 256, 0, stream>>>(conv_w, aF);
    conv_pool<<<NB, 256, 0, stream>>>(x, aF, Vp);
    heads_kernel<<<NB / 8, 256, 0, stream>>>(Vp,
        fc1_w, fc1_b, fc2_w, fc2_b, fc3_w, fc3_b, sc_w, sc_b,
        fc4_w, fc4_b, fc5_w, fc5_b, (float*)d_out);
}
